// FromReference_86766929314314
// MI455X (gfx1250) — compile-verified, matrix operand order checked by reading, not run on hardware
//
#include <hip/hip_runtime.h>

typedef __attribute__((ext_vector_type(16))) __bf16 v16bf;
typedef __attribute__((ext_vector_type(8)))  __bf16 v8bf;
typedef __attribute__((ext_vector_type(8)))  float  v8f;

// Compiler-only fence: prevents store-to-load forwarding across the LDS
// cross-lane relayout (HW keeps same-wave LDS ops in order, so this is free).
#define LDS_RELAYOUT_FENCE() asm volatile("" ::: "memory")

// gelu (tanh approx, matching jax.nn.gelu approximate=True) via HW exp2/rcp:
// tanh(y) = 1 - 2/(exp2(2*log2e*y)+1)
static __device__ __forceinline__ float gelu_f(float x) {
    float y  = 0.7978845608028654f * (x + 0.044715f * x * x * x);
    float e  = __builtin_amdgcn_exp2f(y * 2.8853900817779268f);
    float th = 1.0f - 2.0f * __builtin_amdgcn_rcpf(e + 1.0f);
    return 0.5f * x * (1.0f + th);
}

static __device__ __forceinline__ v8f wmma_bf16(v16bf a, v16bf b, v8f c) {
    return __builtin_amdgcn_wmma_f32_16x16x32_bf16(false, a, false, b, (short)0, c, false, false);
}

// B=4, S=64, C_IN=8, R=64, N=65536, SAMP=12, HID=64, C_OUT=8, c_mlp_in=11
__global__ __launch_bounds__(256) void mlp16_kernel(
    const float* __restrict__ ref,      // (4,64,8,64)
    const float* __restrict__ pc,       // (2,65536)
    const int*   __restrict__ lookup,   // (65536)
    const int*   __restrict__ samp,     // (65536,12)
    const float* __restrict__ W1, const float* __restrict__ b1,
    const float* __restrict__ W2, const float* __restrict__ b2,
    const float* __restrict__ W3, const float* __restrict__ b3,
    float* __restrict__ out)            // (4,8,65536)
{
    __shared__ __align__(16) __bf16 ldsbuf[8][16 * 64];   // 2 KB scratch per wave

    const int lane  = threadIdx.x & 31;
    const int wave  = threadIdx.x >> 5;
    const int task  = blockIdx.x * 8 + wave;   // 0..16383
    const int bIdx  = task >> 12;              // batch 0..3
    const int n0    = (task & 4095) * 16;      // first pixel of this wave's tile

    const int col   = lane & 15;               // matrix column / tile row index
    const int khalf = lane >> 4;               // which K-half this lane holds
    const int n     = n0 + col;                // pixel owned by this lane

    __bf16* myLds = &ldsbuf[wave][0];

    // ---- loop-invariant per-pixel data ----
    const int    sub  = lookup[n];
    const float  p0   = pc[n];
    const float  p1   = pc[65536 + n];
    const float* refB = ref + (bIdx * 64 + sub) * 8 * 64;   // + c*64 + r

    int sIdx[12];
    {
        const int4* sp = reinterpret_cast<const int4*>(samp + n * 12);
        int4 a = sp[0], b = sp[1], c = sp[2];
        sIdx[0]=a.x; sIdx[1]=a.y; sIdx[2]=a.z;  sIdx[3]=a.w;
        sIdx[4]=b.x; sIdx[5]=b.y; sIdx[6]=b.z;  sIdx[7]=b.w;
        sIdx[8]=c.x; sIdx[9]=c.y; sIdx[10]=c.z; sIdx[11]=c.w;
    }

    // ---- weight fragments (WMMA A operands), built once, kept in VGPRs ----
    // A 16x32 bf16 layout: lane<16 holds K=0..7 (elems 0..7) and K=16..23 (elems 8..15);
    //                      lane>=16 holds K=8..15 and K=24..31. Row m = 16*t + col.
    v16bf w1f[4], w3f[2], w2f[4][2];
    v8f   b2f[4];

    #pragma unroll
    for (int t = 0; t < 4; ++t) {
        const int m = t * 16 + col;
        v16bf f;
        #pragma unroll
        for (int i = 0; i < 16; ++i) f[i] = (__bf16)0.0f;
        #pragma unroll
        for (int i = 0; i < 8; ++i) {
            const int k = khalf * 8 + i;      // feature index 0..15 (K padded to 32)
            float v = 0.0f;
            if (k < 11)       v = W1[m * 11 + k];
            else if (k == 11) v = b1[m];      // bias folded: input feature K=11 == 1.0
            f[i] = (__bf16)v;
        }
        w1f[t] = f;

        #pragma unroll
        for (int q = 0; q < 2; ++q) {
            v16bf g;
            #pragma unroll
            for (int i = 0; i < 8; ++i) {
                g[i]     = (__bf16)W2[m * 64 + q * 32 + khalf * 8 + i];
                g[i + 8] = (__bf16)W2[m * 64 + q * 32 + 16 + khalf * 8 + i];
            }
            w2f[t][q] = g;
        }

        v8f bb;   // b2 as initial C fragment: reg r -> row 16t + 8*khalf + r
        #pragma unroll
        for (int r = 0; r < 8; ++r) bb[r] = b2[t * 16 + khalf * 8 + r];
        b2f[t] = bb;
    }

    #pragma unroll
    for (int q = 0; q < 2; ++q) {     // W3: rows 0..7 real, 8..15 zero-padded
        v16bf g;
        #pragma unroll
        for (int i = 0; i < 8; ++i) {
            float a = (col < 8) ? W3[col * 64 + q * 32 + khalf * 8 + i]      : 0.0f;
            float c = (col < 8) ? W3[col * 64 + q * 32 + 16 + khalf * 8 + i] : 0.0f;
            g[i]     = (__bf16)a;
            g[i + 8] = (__bf16)c;
        }
        w3f[q] = g;
    }

    v8f outAcc;
    #pragma unroll
    for (int r = 0; r < 8; ++r) outAcc[r] = 0.0f;

    for (int s = 0; s < 12; ++s) {
        const int rs = sIdx[s];

        // ---- input fragment (B operand 32x16): feature K per lane-half ----
        v16bf xf;
        #pragma unroll
        for (int i = 0; i < 16; ++i) xf[i] = (__bf16)0.0f;
        if (khalf == 0) {                         // K = 0..7
            xf[0] = (__bf16)(-1.0f + (float)rs * (2.0f / 63.0f));  // rc[rs]
            xf[1] = (__bf16)p0;
            xf[2] = (__bf16)p1;
            xf[3] = (__bf16)refB[0 * 64 + rs];
            xf[4] = (__bf16)refB[1 * 64 + rs];
            xf[5] = (__bf16)refB[2 * 64 + rs];
            xf[6] = (__bf16)refB[3 * 64 + rs];
            xf[7] = (__bf16)refB[4 * 64 + rs];
        } else {                                  // K = 8..15
            xf[0] = (__bf16)refB[5 * 64 + rs];
            xf[1] = (__bf16)refB[6 * 64 + rs];
            xf[2] = (__bf16)refB[7 * 64 + rs];
            xf[3] = (__bf16)1.0f;                 // bias-carrier feature (K=11)
        }

        // ---- layer 1: h1 = gelu(W1 x + b1) -> LDS (bf16, [col][h]) ----
        #pragma unroll
        for (int t = 0; t < 4; ++t) {
            v8f c;
            #pragma unroll
            for (int r = 0; r < 8; ++r) c[r] = 0.0f;
            c = wmma_bf16(w1f[t], xf, c);
            v8bf pk;
            #pragma unroll
            for (int r = 0; r < 8; ++r) pk[r] = (__bf16)gelu_f(c[r]);
            *reinterpret_cast<v8bf*>(myLds + col * 64 + t * 16 + khalf * 8) = pk;
        }
        LDS_RELAYOUT_FENCE();   // cross-lane relayout: forbid store->load forwarding

        // ---- reload h1 as B fragments (ds_load_b128 x4, in-order within wave) ----
        v16bf h1f[2];
        #pragma unroll
        for (int q = 0; q < 2; ++q) {
            v8bf lo = *reinterpret_cast<const v8bf*>(myLds + col * 64 + q * 32 + khalf * 8);
            v8bf hi = *reinterpret_cast<const v8bf*>(myLds + col * 64 + q * 32 + 16 + khalf * 8);
            h1f[q] = __builtin_shufflevector(lo, hi, 0,1,2,3,4,5,6,7,8,9,10,11,12,13,14,15);
        }
        LDS_RELAYOUT_FENCE();   // keep the next stores after these loads

        // ---- layer 2: h2 = gelu(W2 h1 + b2), b2 as initial C ----
        #pragma unroll
        for (int t = 0; t < 4; ++t) {
            v8f c = b2f[t];
            c = wmma_bf16(w2f[t][0], h1f[0], c);
            c = wmma_bf16(w2f[t][1], h1f[1], c);
            v8bf pk;
            #pragma unroll
            for (int r = 0; r < 8; ++r) pk[r] = (__bf16)gelu_f(c[r]);
            *reinterpret_cast<v8bf*>(myLds + col * 64 + t * 16 + khalf * 8) = pk;
        }
        LDS_RELAYOUT_FENCE();   // cross-lane relayout: forbid store->load forwarding

        // ---- layer 3, chained into the cross-sample accumulator ----
        v16bf h2f[2];
        #pragma unroll
        for (int q = 0; q < 2; ++q) {
            v8bf lo = *reinterpret_cast<const v8bf*>(myLds + col * 64 + q * 32 + khalf * 8);
            v8bf hi = *reinterpret_cast<const v8bf*>(myLds + col * 64 + q * 32 + 16 + khalf * 8);
            h2f[q] = __builtin_shufflevector(lo, hi, 0,1,2,3,4,5,6,7,8,9,10,11,12,13,14,15);
        }
        LDS_RELAYOUT_FENCE();   // keep next iteration's stores after these loads

        outAcc = wmma_bf16(w3f[0], h2f[0], outAcc);
        outAcc = wmma_bf16(w3f[1], h2f[1], outAcc);
    }

    // ---- mean over 12 samples + b3; lanes 0..15 hold rows 0..7 (= C_OUT) ----
    if (khalf == 0) {
        #pragma unroll
        for (int r = 0; r < 8; ++r) {
            out[(bIdx * 8 + r) * 65536 + n] = outAcc[r] * (1.0f / 12.0f) + b3[r];
        }
    }
}

extern "C" void kernel_launch(void* const* d_in, const int* in_sizes, int n_in,
                              void* d_out, int out_size, void* d_ws, size_t ws_size,
                              hipStream_t stream) {
    const float* ref  = (const float*)d_in[0];
    const float* pc   = (const float*)d_in[1];
    const int*   lk   = (const int*)d_in[2];
    const int*   smp  = (const int*)d_in[3];
    const float* W1   = (const float*)d_in[4];
    const float* b1   = (const float*)d_in[5];
    const float* W2   = (const float*)d_in[6];
    const float* b2   = (const float*)d_in[7];
    const float* W3   = (const float*)d_in[8];
    const float* b3   = (const float*)d_in[9];
    float* out = (float*)d_out;

    // 16384 wave-tasks (4 batches x 4096 pixel tiles), 8 waves per 256-thread block
    dim3 grid(2048), block(256);
    hipLaunchKernelGGL(mlp16_kernel, grid, block, 0, stream,
                       ref, pc, lk, smp, W1, b1, W2, b2, W3, b3, out);
}
